// SemLA_36833639531250
// MI455X (gfx1250) — matrix operand order repaired and checked by reading, not compiled, not run on hardware
//
#include <hip/hip_runtime.h>
#include <hip/hip_bf16.h>
#include <cstdint>

// ---------------------------------------------------------------------------
// SemLA mutual-top-1 matcher for MI455X (gfx1250, wave32, WMMA + async-LDS)
// ---------------------------------------------------------------------------
// H=96 W=128 C=256 L=12288.  conf = (fv/sqrt(C)) @ (fi/sqrt(C))^T / 0.1
// f16 WMMA GEMM, two passes (max-reduce, then match) — the 604MB conf matrix
// is never materialized. B tiles are staged to LDS with gfx1250 async
// global->LDS copies and shared by all 8 waves; each wave computes a 32x64
// strip (8 WMMA accumulators) so each LDS B fragment feeds two WMMAs.
// ---------------------------------------------------------------------------

#define LL   12288
#define CC   256
#define HH   96
#define WW   128
#define UPH  768
#define UPW  1024
#define NEGF (-3.0e38f)
#define BPAD 264   // B tile row stride in halves (528B = 33*16B, keeps 16B align)

typedef __attribute__((ext_vector_type(16))) _Float16 v16h;
typedef __attribute__((ext_vector_type(8)))  _Float16 v8h;
typedef __attribute__((ext_vector_type(8)))  float    v8f;

union Frag16 { v16h v; v8h h[2]; };

__device__ __forceinline__ void atomicMaxF(float* addr, float val) {
    int* ai = (int*)addr;
    int cur = __float_as_int(*addr);
    while (__int_as_float(cur) < val) {
        int prev = atomicCAS(ai, cur, __float_as_int(val));
        if (prev == cur) break;
        cur = prev;
    }
}

// ----- convert [C,L] f32 -> [L,C] f16 (scaled), via padded LDS transpose ----
__global__ __launch_bounds__(256) void semla_convert(const float* __restrict__ src,
                                                     _Float16* __restrict__ dst) {
    __shared__ float tile[32][33];
    const int l0 = blockIdx.x * 32;
    const int c0 = blockIdx.y * 32;
    const int tid = threadIdx.x;
#pragma unroll
    for (int i = 0; i < 4; ++i) {
        int idx = tid + i * 256;
        int cc = idx >> 5, ll = idx & 31;
        tile[ll][cc] = src[(size_t)(c0 + cc) * LL + (l0 + ll)];
    }
    __syncthreads();
    const float s = 0.19764235376052372f; // sqrt(10)/16 = sqrt(1/0.1)/sqrt(C)
#pragma unroll
    for (int i = 0; i < 4; ++i) {
        int idx = tid + i * 256;
        int ll = idx >> 5, cc = idx & 31;
        dst[(size_t)(l0 + ll) * CC + (c0 + cc)] = (_Float16)(tile[ll][cc] * s);
    }
}

// ----- init reduction buffers ----------------------------------------------
__global__ __launch_bounds__(256) void semla_init(float* __restrict__ row_max,
                                                  float* __restrict__ col_max,
                                                  int* __restrict__ j_best) {
    int i = blockIdx.x * 256 + threadIdx.x;
    if (i < LL) {
        row_max[i] = NEGF;
        col_max[i] = NEGF;
        j_best[i]  = 0x7fffffff;
    }
}

// ----- WMMA GEMM, two passes -----------------------------------------------
// Block = 256 threads = 8 waves. Block tile: 256(M) x 64(N).
// Wave w owns rows [bx*256 + w*32, +32) as 2 M-tiles x 4 N-tiles (8 accs).
// B tile (64x256 f16) staged once per block into LDS with async copies.
template <int PASS>
__global__ __launch_bounds__(256) void semla_gemm(const _Float16* __restrict__ fv,
                                                  const _Float16* __restrict__ fi,
                                                  const float* __restrict__ sa_vi,
                                                  const float* __restrict__ sa_ir,
                                                  float* __restrict__ row_max,
                                                  float* __restrict__ col_max,
                                                  int* __restrict__ j_best) {
    __shared__ _Float16 lds_b[64 * BPAD];

    const int tid  = threadIdx.x;
    const int lane = tid & 31;
    const int wave = tid >> 5;
    const int half = lane >> 4;   // 0: lanes 0-15, 1: lanes 16-31
    const int l16  = lane & 15;

    const int m_base = blockIdx.x * 256 + wave * 32;
    const int n_base = blockIdx.y * 64;

    // ---- stage B tile 64 rows x 256 halves into LDS (async global->LDS) ----
    // 64*512B = 32KB; 256 threads x 8 chunks of 16B each.
#pragma unroll
    for (int i = 0; i < 8; ++i) {
        int idx = tid + i * 256;           // 0..2047
        int row = idx >> 5;                // 0..63
        int c16 = idx & 31;                // 16B chunk within row
        uint32_t lds_off = (uint32_t)(row * (BPAD * 2) + c16 * 16);
        uint64_t gaddr = (uint64_t)(uintptr_t)(fi + (size_t)(n_base + row) * CC + c16 * 8);
        asm volatile("global_load_async_to_lds_b128 %0, %1, off"
                     :: "v"(lds_off), "v"(gaddr) : "memory");
    }
    asm volatile("s_wait_asynccnt 0x0" ::: "memory");
    __syncthreads();

    v8f acc0[4] = {};
    v8f acc1[4] = {};

    for (int k0 = 0; k0 < CC; k0 += 32) {
        // A fragments: 16x32 f16. Lane half 0: K=k0..k0+7 / k0+16..k0+23,
        // lane half 1: K=k0+8..k0+15 / k0+24..k0+31 (ISA VGPR layout).
        const _Float16* ap0 = fv + (size_t)(m_base + l16) * CC + k0 + half * 8;
        const _Float16* ap1 = ap0 + (size_t)16 * CC;
        Frag16 a0, a1;
        a0.h[0] = *(const v8h*)(ap0);
        a0.h[1] = *(const v8h*)(ap0 + 16);
        a1.h[0] = *(const v8h*)(ap1);
        a1.h[1] = *(const v8h*)(ap1 + 16);
#pragma unroll
        for (int t = 0; t < 4; ++t) {
            // B fragment: 32x16 f16, N = lane%16, K split per lane half like A.
            const _Float16* bp = &lds_b[(t * 16 + l16) * BPAD + k0 + half * 8];
            Frag16 b;
            b.h[0] = *(const v8h*)(bp);
            b.h[1] = *(const v8h*)(bp + 16);
            acc0[t] = __builtin_amdgcn_wmma_f32_16x16x32_f16(
                false, a0.v, false, b.v, (short)0, acc0[t], false, false);
            acc1[t] = __builtin_amdgcn_wmma_f32_16x16x32_f16(
                false, a1.v, false, b.v, (short)0, acc1[t], false, false);
        }
    }

    // Row validity (acc element v -> row v + half*8 within its 16-row tile)
    bool validi0[8], validi1[8];
#pragma unroll
    for (int v = 0; v < 8; ++v) {
        validi0[v] = sa_vi[m_base + half * 8 + v] > 0.5f;
        validi1[v] = sa_vi[m_base + 16 + half * 8 + v] > 0.5f;
    }

    if (PASS == 1) {
        float rmax0[8], rmax1[8];
#pragma unroll
        for (int v = 0; v < 8; ++v) { rmax0[v] = NEGF; rmax1[v] = NEGF; }
#pragma unroll
        for (int t = 0; t < 4; ++t) {
            const int gj = n_base + t * 16 + l16;
            const bool vj = sa_ir[gj] > 0.5f;
            float cmax = NEGF;
#pragma unroll
            for (int v = 0; v < 8; ++v) {
                float c0 = acc0[t][v];
                float c1 = acc1[t][v];
                float m0 = (vj && validi0[v]) ? c0 : NEGF;
                float m1 = (vj && validi1[v]) ? c1 : NEGF;
                rmax0[v] = fmaxf(rmax0[v], m0);
                rmax1[v] = fmaxf(rmax1[v], m1);
                cmax = fmaxf(cmax, fmaxf(m0, m1));
            }
            cmax = fmaxf(cmax, __shfl_xor(cmax, 16, 32)); // fold lane halves
            if (half == 0 && cmax > NEGF) atomicMaxF(&col_max[gj], cmax);
        }
        // reduce each row across the 16 lanes of this half-wave
#pragma unroll
        for (int v = 0; v < 8; ++v) {
            float r0 = rmax0[v], r1 = rmax1[v];
            r0 = fmaxf(r0, __shfl_xor(r0, 1, 32));
            r0 = fmaxf(r0, __shfl_xor(r0, 2, 32));
            r0 = fmaxf(r0, __shfl_xor(r0, 4, 32));
            r0 = fmaxf(r0, __shfl_xor(r0, 8, 32));
            r1 = fmaxf(r1, __shfl_xor(r1, 1, 32));
            r1 = fmaxf(r1, __shfl_xor(r1, 2, 32));
            r1 = fmaxf(r1, __shfl_xor(r1, 4, 32));
            r1 = fmaxf(r1, __shfl_xor(r1, 8, 32));
            if (l16 == 0) {
                if (r0 > NEGF) atomicMaxF(&row_max[m_base + half * 8 + v], r0);
                if (r1 > NEGF) atomicMaxF(&row_max[m_base + 16 + half * 8 + v], r1);
            }
        }
    } else {
        float rmv0[8], rmv1[8];
#pragma unroll
        for (int v = 0; v < 8; ++v) {
            rmv0[v] = row_max[m_base + half * 8 + v];
            rmv1[v] = row_max[m_base + 16 + half * 8 + v];
        }
#pragma unroll
        for (int t = 0; t < 4; ++t) {
            const int gj = n_base + t * 16 + l16;
            const bool vj = sa_ir[gj] > 0.5f;
            const float cm = col_max[gj];
#pragma unroll
            for (int v = 0; v < 8; ++v) {
                float c0 = acc0[t][v];
                float c1 = acc1[t][v];
                if (vj && validi0[v] && c0 > 0.0f && c0 == rmv0[v] && c0 == cm)
                    atomicMin(&j_best[m_base + half * 8 + v], gj);
                if (vj && validi1[v] && c1 > 0.0f && c1 == rmv1[v] && c1 == cm)
                    atomicMin(&j_best[m_base + 16 + half * 8 + v], gj);
            }
        }
    }
}

// ----- 8x bilinear upsample, align_corners=True ----------------------------
__global__ __launch_bounds__(256) void semla_upsample(const float* __restrict__ src,
                                                      float* __restrict__ dst) {
    int idx = blockIdx.x * 256 + threadIdx.x;
    if (idx >= UPH * UPW) return;
    int ox = idx & (UPW - 1);
    int oy = idx >> 10;
    float px = ox * (float)(WW - 1) / (float)(UPW - 1);
    float py = oy * (float)(HH - 1) / (float)(UPH - 1);
    int x0 = (int)px, y0 = (int)py;
    float fx = px - (float)x0, fy = py - (float)y0;
    int x1 = min(x0 + 1, WW - 1), y1 = min(y0 + 1, HH - 1);
    float v00 = src[y0 * WW + x0], v01 = src[y0 * WW + x1];
    float v10 = src[y1 * WW + x0], v11 = src[y1 * WW + x1];
    dst[idx] = (1.0f - fy) * ((1.0f - fx) * v00 + fx * v01) +
               fy * ((1.0f - fx) * v10 + fx * v11);
}

// ----- final writeout: mkpts0, mkpts1, mask_v, passthroughs ----------------
__global__ __launch_bounds__(256) void semla_writeout(const int* __restrict__ j_best,
                                                      const float* __restrict__ sa_vi,
                                                      const float* __restrict__ sa_ir,
                                                      float* __restrict__ out) {
    int i = blockIdx.x * 256 + threadIdx.x;
    if (i >= LL) return;
    const int O_MK0 = 0, O_MK1 = 24576, O_MASK = 49152, O_SAVI = 61440, O_SAIR = 73728;
    out[O_MK0 + 2 * i]     = (float)((i % WW) * 8);
    out[O_MK0 + 2 * i + 1] = (float)((i / WW) * 8);
    int j = j_best[i];
    bool matched = (j != 0x7fffffff);
    int jj = matched ? j : 0;
    out[O_MK1 + 2 * i]     = (float)((jj % WW) * 8);
    out[O_MK1 + 2 * i + 1] = (float)((jj / WW) * 8);
    out[O_MASK + i] = matched ? 1.0f : 0.0f;
    out[O_SAVI + i] = sa_vi[i];
    out[O_SAIR + i] = sa_ir[i];
}

extern "C" void kernel_launch(void* const* d_in, const int* in_sizes, int n_in,
                              void* d_out, int out_size, void* d_ws, size_t ws_size,
                              hipStream_t stream) {
    const float* feat_reg_vi = (const float*)d_in[0]; // [C,L]
    const float* feat_reg_ir = (const float*)d_in[1]; // [C,L]
    const float* feat_sa_vi  = (const float*)d_in[2]; // [L]
    const float* feat_sa_ir  = (const float*)d_in[3]; // [L]
    float* out = (float*)d_out;

    // workspace layout (bytes)
    char* ws = (char*)d_ws;
    _Float16* fv_h    = (_Float16*)(ws);                         // L*C*2 = 6,291,456
    _Float16* fi_h    = (_Float16*)(ws + 6291456);               // L*C*2
    float*    row_max = (float*)   (ws + 12582912);              // L*4
    float*    col_max = (float*)   (ws + 12632064);              // L*4
    int*      j_best  = (int*)     (ws + 12681216);              // L*4

    // 1) convert + scale + transpose to f16 [L,C]
    dim3 cgrid(LL / 32, CC / 32);
    semla_convert<<<cgrid, 256, 0, stream>>>(feat_reg_vi, fv_h);
    semla_convert<<<cgrid, 256, 0, stream>>>(feat_reg_ir, fi_h);

    // 2) init reductions
    semla_init<<<LL / 256, 256, 0, stream>>>(row_max, col_max, j_best);

    // 3) pass 1: row/col maxima over valid entries
    dim3 ggrid(LL / 256, LL / 64);
    semla_gemm<1><<<ggrid, 256, 0, stream>>>(fv_h, fi_h, feat_sa_vi, feat_sa_ir,
                                             row_max, col_max, j_best);
    // 4) pass 2: mutual top-1 match (bitwise-identical recompute)
    semla_gemm<2><<<ggrid, 256, 0, stream>>>(fv_h, fi_h, feat_sa_vi, feat_sa_ir,
                                             row_max, col_max, j_best);

    // 5) upsamples (output order: sa_ir_up first, then sa_vi_up)
    const int O_IRUP = 86016, O_VIUP = 872448;
    semla_upsample<<<(UPH * UPW) / 256, 256, 0, stream>>>(feat_sa_ir, out + O_IRUP);
    semla_upsample<<<(UPH * UPW) / 256, 256, 0, stream>>>(feat_sa_vi, out + O_VIUP);

    // 6) keypoints / mask / passthrough
    semla_writeout<<<LL / 256, 256, 0, stream>>>(j_best, feat_sa_vi, feat_sa_ir, out);
}
